// PNA_46574625358332
// MI455X (gfx1250) — matrix operand order repaired and checked by reading, not compiled, not run on hardware
//
#include <hip/hip_runtime.h>
#include <hip/hip_bf16.h>

typedef __bf16 bf16_t;
typedef __attribute__((ext_vector_type(8)))  __bf16 v8bf;
typedef __attribute__((ext_vector_type(16))) __bf16 v16bf;
typedef __attribute__((ext_vector_type(8)))  float  v8f;

#define HD 128
#define NTILES 8

// ---------- helpers ----------
__device__ __forceinline__ bf16_t f2bf(float f) {
  unsigned u = __float_as_uint(f);
  unsigned r = u + 0x7FFFu + ((u >> 16) & 1u);   // RNE
  unsigned short h = (unsigned short)(r >> 16);
  return __builtin_bit_cast(bf16_t, h);
}
// order-preserving float<->uint for integer atomicMin/Max
__device__ __forceinline__ unsigned fenc(float f) {
  unsigned u = __float_as_uint(f);
  return (u & 0x80000000u) ? ~u : (u | 0x80000000u);
}
__device__ __forceinline__ float fdec(unsigned u) {
  unsigned v = (u & 0x80000000u) ? (u & 0x7FFFFFFFu) : ~u;
  return __uint_as_float(v);
}

// A fragment (16x32 bf16) from LDS row-major tile.
// lanes 0-15: K = k0+0..7 and k0+16..23 ; lanes 16-31: K = k0+8..15 and k0+24..31
__device__ __forceinline__ v16bf load_a(const bf16_t* row, int k0, int half) {
  v8bf lo = *(const v8bf*)(row + k0 + half * 8);
  v8bf hi = *(const v8bf*)(row + k0 + 16 + half * 8);
  v16bf a;
#pragma unroll
  for (int i = 0; i < 8; ++i) { a[i] = lo[i]; a[i + 8] = hi[i]; }
  return a;
}
// B fragment from pre-packed weights: ((kc*8+nt)*32+lane)*16 contiguous bf16
__device__ __forceinline__ v16bf load_b(const bf16_t* Wp, int kc, int nt, int lane) {
  return *(const v16bf*)(Wp + ((((kc * NTILES) + nt) * 32 + lane) << 4));
}
__device__ __forceinline__ v8f wmma_bf16(v16bf a, v16bf b, v8f c) {
  return __builtin_amdgcn_wmma_f32_16x16x32_bf16(false, a, false, b, (short)0, c,
                                                 false, false);
}

// ---------- small utility kernels ----------
__global__ void k_fill_u32(unsigned* p, unsigned v, long n) {
  long i = (long)blockIdx.x * blockDim.x + threadIdx.x;
  if (i < n) p[i] = v;
}

// pack fp32 weight [K x 128] into B-fragment layout (bf16).
// mode 1: em1 reorder (tile rows 0-127 <-> source rows 128-255)
__global__ void k_pack(const float* __restrict__ W, bf16_t* __restrict__ out,
                       int K, int mode) {
  int t = blockIdx.x * blockDim.x + threadIdx.x;
  if (t >= K * HD) return;
  int i = t & 15, lane = (t >> 4) & 31, nt = (t >> 9) & 7, kc = t >> 12;
  int klocal = ((i >> 3) << 4) + ((lane >> 4) << 3) + (i & 7);           // 0..31
  int krow = kc * 32 + klocal;
  int n = (nt << 4) + (lane & 15);
  int srow = krow;
  if (mode == 1) srow = (krow < 128) ? krow + 128 : ((krow < 256) ? krow - 128 : krow);
  out[t] = f2bf(W[(long)srow * HD + n]);
}

// Wc = post_W(1664x128) @ lin_W(128x128); bc = post_b@lin_W + lin_b
__global__ void k_combine(const float* __restrict__ postW, const float* __restrict__ postb,
                          const float* __restrict__ linW, const float* __restrict__ linb,
                          float* __restrict__ Wc, float* __restrict__ bc) {
  long idx = (long)blockIdx.x * blockDim.x + threadIdx.x;
  if (idx >= (long)1664 * HD) return;
  int r = (int)(idx >> 7), c = (int)(idx & 127);
  float s = 0.f;
  for (int k = 0; k < HD; ++k) s += postW[(long)r * HD + k] * linW[(long)k * HD + c];
  Wc[idx] = s;
  if (r == 0) {
    float sb = 0.f;
    for (int k = 0; k < HD; ++k) sb += postb[k] * linW[(long)k * HD + c];
    bc[c] = sb + linb[c];
  }
}

__global__ void k_degree(const int* __restrict__ dst, float* __restrict__ cnt, int E) {
  int e = blockIdx.x * blockDim.x + threadIdx.x;
  if (e < E) atomicAdd(&cnt[dst[e]], 1.0f);
}

__global__ void k_logsum(const float* __restrict__ cnt, float* __restrict__ sumlog, int N) {
  __shared__ float sh[256];
  int i = blockIdx.x * blockDim.x + threadIdx.x;
  sh[threadIdx.x] = (i < N) ? logf(cnt[i] + 1.0f) : 0.0f;
  __syncthreads();
  for (int s = blockDim.x / 2; s > 0; s >>= 1) {
    if ((int)threadIdx.x < s) sh[threadIdx.x] += sh[threadIdx.x + s];
    __syncthreads();
  }
  if (threadIdx.x == 0) atomicAdd(sumlog, sh[0]);
}

// ---------- generic embedding GEMM: in(rows x K) @ Wp -> out(rows x 128)+bias ----------
__global__ __launch_bounds__(128) void k_embed(const float* __restrict__ in, int K,
                                               const bf16_t* __restrict__ Wp,
                                               const float* __restrict__ bias,
                                               float* __restrict__ outf,
                                               bf16_t* __restrict__ outb, int rows) {
  __shared__ __align__(16) bf16_t tile[64 * 72];   // max K=64, ldk=K+8
  const int ldk = K + 8;
  const int tid = threadIdx.x;
  const long base = (long)blockIdx.x * 64;
  const int gk = K >> 3;
  for (int g = tid; g < 64 * gk; g += 128) {
    int r = g / gk, kg = g % gk;
    long e = base + r;
    v8bf v;
    if (e < rows) {
      const float* s = in + e * K + kg * 8;
#pragma unroll
      for (int i = 0; i < 8; ++i) v[i] = f2bf(s[i]);
    } else {
#pragma unroll
      for (int i = 0; i < 8; ++i) v[i] = f2bf(0.f);
    }
    *(v8bf*)(tile + r * ldk + kg * 8) = v;
  }
  __syncthreads();
  const int lane = tid & 31, wave = tid >> 5;
  const int half = lane >> 4, rsel = lane & 15;
  const bf16_t* arow = tile + (wave * 16 + rsel) * ldk;
  v8f zero = {0.f, 0.f, 0.f, 0.f, 0.f, 0.f, 0.f, 0.f};
  v8f acc[NTILES];
#pragma unroll
  for (int t = 0; t < NTILES; ++t) acc[t] = zero;
  const int nkc = K >> 5;
  for (int kc = 0; kc < nkc; ++kc) {
    v16bf a = load_a(arow, kc * 32, half);
#pragma unroll
    for (int t = 0; t < NTILES; ++t) acc[t] = wmma_bf16(a, load_b(Wp, kc, t, lane), acc[t]);
  }
#pragma unroll
  for (int t = 0; t < NTILES; ++t) {
    int n = t * 16 + rsel;
    float bv = bias[n];
#pragma unroll
    for (int i = 0; i < 8; ++i) {
      long e = base + wave * 16 + i + half * 8;
      if (e < rows) {
        float v = acc[t][i] + bv;
        outf[e * HD + n] = v;
        if (outb) outb[e * HD + n] = f2bf(v);
      }
    }
  }
}

// ---------- stage [x_dst | x_src | ea] edge tile into LDS (ldk=392) ----------
__device__ __forceinline__ void stage_edge(bf16_t* tile, int rows, long base,
                                           const bf16_t* __restrict__ xb,
                                           const float* __restrict__ ea,
                                           const int* __restrict__ srcp,
                                           const int* __restrict__ dstp, int E,
                                           int nthreads) {
  for (int g = threadIdx.x; g < rows * 48; g += nthreads) {
    int r = g / 48, kg = g % 48;
    long e = base + r;
    if (e >= E) e = E - 1;
    v8bf v;
    if (kg < 16) {
      v = *(const v8bf*)(xb + (long)dstp[e] * HD + kg * 8);
    } else if (kg < 32) {
      v = *(const v8bf*)(xb + (long)srcp[e] * HD + (kg - 16) * 8);
    } else {
      const float* s = ea + e * HD + (kg - 32) * 8;
#pragma unroll
      for (int i = 0; i < 8; ++i) v[i] = f2bf(s[i]);
    }
    *(v8bf*)(tile + r * 392 + kg * 8) = v;
  }
}

// ---------- edge message GEMM + fused segment stats ----------
__global__ __launch_bounds__(128) void k_edge_msg(
    const bf16_t* __restrict__ xb, const float* __restrict__ ea,
    const int* __restrict__ eidx, const bf16_t* __restrict__ Wp,
    const float* __restrict__ bias, float* __restrict__ sum,
    float* __restrict__ sumsq, unsigned* __restrict__ mnb,
    unsigned* __restrict__ mxb, int E) {
  __shared__ __align__(16) bf16_t tile[64 * 392];
  const int* srcp = eidx;
  const int* dstp = eidx + E;
  const long base = (long)blockIdx.x * 64;
  stage_edge(tile, 64, base, xb, ea, srcp, dstp, E, 128);
  __syncthreads();
  const int tid = threadIdx.x, lane = tid & 31, wave = tid >> 5;
  const int half = lane >> 4, rsel = lane & 15;
  const bf16_t* arow = tile + (wave * 16 + rsel) * 392;
  v8f zero = {0.f, 0.f, 0.f, 0.f, 0.f, 0.f, 0.f, 0.f};
  v8f acc[NTILES];
#pragma unroll
  for (int t = 0; t < NTILES; ++t) acc[t] = zero;
  for (int kc = 0; kc < 12; ++kc) {
    v16bf a = load_a(arow, kc * 32, half);
#pragma unroll
    for (int t = 0; t < NTILES; ++t) acc[t] = wmma_bf16(a, load_b(Wp, kc, t, lane), acc[t]);
  }
#pragma unroll
  for (int t = 0; t < NTILES; ++t) {
    int n = t * 16 + rsel;
    float bv = bias[n];
#pragma unroll
    for (int i = 0; i < 8; ++i) {
      long e = base + wave * 16 + i + half * 8;
      if (e < E) {
        int d = dstp[e];
        float v = acc[t][i] + bv;
        long o = (long)d * HD + n;
        atomicAdd(sum + o, v);
        atomicAdd(sumsq + o, v * v);
        atomicMin(mnb + o, fenc(v));
        atomicMax(mxb + o, fenc(v));
      }
    }
  }
}

// ---------- node update: build [x|agg|agg*amp|agg*att] (K=1664) and GEMM ----------
__global__ __launch_bounds__(128) void k_node_update(
    const float* __restrict__ x, const float* __restrict__ sum,
    const float* __restrict__ sumsq, const unsigned* __restrict__ mnb,
    const unsigned* __restrict__ mxb, const float* __restrict__ cnt,
    const float* __restrict__ sumlog, const bf16_t* __restrict__ Wp,
    const float* __restrict__ bc, float* __restrict__ outbuf, int Nn) {
  __shared__ __align__(16) bf16_t tile[16 * 1672];   // ldk = 1664+8
  const int tid = threadIdx.x;
  const long base = (long)blockIdx.x * 16;
  const float avg = sumlog[0] / (float)Nn;
  for (int p = tid; p < 16 * HD; p += 128) {
    int r = p >> 7, f = p & 127;
    long node = base + r;
    bf16_t* L = tile + r * 1672;
    if (node >= Nn) {
#pragma unroll
      for (int blk = 0; blk < 13; ++blk) L[blk * 128 + f] = f2bf(0.f);
    } else {
      long o = node * HD + f;
      float c = cnt[node];
      float cc = fmaxf(c, 1.f);
      float mean = sum[o] / cc;
      float m2 = sumsq[o] / cc;
      float sd = sqrtf(fmaxf(m2 - mean * mean, 0.f) + 1e-5f);
      bool has = c > 0.f;
      float mnv = has ? fdec(mnb[o]) : 0.f;
      float mxv = has ? fdec(mxb[o]) : 0.f;
      float lg = logf(cc + 1.f);
      float amp = lg / avg, att = avg / lg;
      L[f] = f2bf(x[o]);
      L[128 + f] = f2bf(mean);
      L[256 + f] = f2bf(mnv);
      L[384 + f] = f2bf(mxv);
      L[512 + f] = f2bf(sd);
      L[640 + f] = f2bf(mean * amp);
      L[768 + f] = f2bf(mnv * amp);
      L[896 + f] = f2bf(mxv * amp);
      L[1024 + f] = f2bf(sd * amp);
      L[1152 + f] = f2bf(mean * att);
      L[1280 + f] = f2bf(mnv * att);
      L[1408 + f] = f2bf(mxv * att);
      L[1536 + f] = f2bf(sd * att);
    }
  }
  __syncthreads();
  const int lane = tid & 31, wave = tid >> 5;
  const int half = lane >> 4, rsel = lane & 15;
  const bf16_t* arow = tile + rsel * 1672;
  v8f zero = {0.f, 0.f, 0.f, 0.f, 0.f, 0.f, 0.f, 0.f};
  v8f a0 = zero, a1 = zero;
  const int t0 = wave * 2, t1 = wave * 2 + 1;
  for (int kc = 0; kc < 52; ++kc) {
    v16bf a = load_a(arow, kc * 32, half);
    a0 = wmma_bf16(a, load_b(Wp, kc, t0, lane), a0);
    a1 = wmma_bf16(a, load_b(Wp, kc, t1, lane), a1);
  }
#pragma unroll
  for (int sel = 0; sel < 2; ++sel) {
    v8f acc = sel ? a1 : a0;
    int t = sel ? t1 : t0;
    int n = t * 16 + rsel;
    float bv = bc[n];
#pragma unroll
    for (int i = 0; i < 8; ++i) {
      long node = base + i + half * 8;
      if (node < Nn) outbuf[node * HD + n] = acc[i] + bv;
    }
  }
}

// ---------- batch-norm stats / apply ----------
__global__ void k_bnstat(const float* __restrict__ out, float* __restrict__ musum,
                         float* __restrict__ varsum, int Nn) {
  __shared__ float ss[128], ss2[128];
  int tid = threadIdx.x;
  if (tid < 128) { ss[tid] = 0.f; ss2[tid] = 0.f; }
  __syncthreads();
  long total = (long)Nn * HD;
  long per = (total + gridDim.x - 1) / gridDim.x;
  long t0 = (long)blockIdx.x * per;
  long t1 = t0 + per; if (t1 > total) t1 = total;
  for (long t = t0 + tid; t < t1; t += blockDim.x) {
    float v = out[t];
    int c = (int)(t & 127);
    atomicAdd(&ss[c], v);
    atomicAdd(&ss2[c], v * v);
  }
  __syncthreads();
  if (tid < 128) { atomicAdd(&musum[tid], ss[tid]); atomicAdd(&varsum[tid], ss2[tid]); }
}

__global__ void k_bnapply(const float* __restrict__ outb, const float* __restrict__ musum,
                          const float* __restrict__ varsum, const float* __restrict__ g,
                          const float* __restrict__ b, float* __restrict__ x,
                          bf16_t* __restrict__ xb, int Nn) {
  long idx = (long)blockIdx.x * blockDim.x + threadIdx.x;
  long total = (long)Nn * HD;
  if (idx >= total) return;
  int c = (int)(idx & 127);
  float mu = musum[c] / (float)Nn;
  float var = varsum[c] / (float)Nn - mu * mu;
  float o = outb[idx];
  float bn = g[c] * (o - mu) * rsqrtf(var + 1e-5f) + b[c];
  float r = fmaxf(bn, 0.f);
  float xn = (x[idx] + r) * 0.5f;
  x[idx] = xn;
  xb[idx] = f2bf(xn);
}

// ---------- fused edge MLP: hid = relu([xd|xs|ea]@W1+b1); ea += (hid@W2+b2)/2 ----------
__global__ __launch_bounds__(64) void k_edge_mlp(
    const bf16_t* __restrict__ xb, float* __restrict__ ea,
    const int* __restrict__ eidx, const bf16_t* __restrict__ W1p,
    const float* __restrict__ b1, const bf16_t* __restrict__ W2p,
    const float* __restrict__ b2, int E) {
  __shared__ __align__(16) bf16_t tile[32 * 392];
  __shared__ __align__(16) bf16_t hid[32 * 136];
  const int* srcp = eidx;
  const int* dstp = eidx + E;
  const long base = (long)blockIdx.x * 32;
  stage_edge(tile, 32, base, xb, ea, srcp, dstp, E, 64);
  __syncthreads();
  const int tid = threadIdx.x, lane = tid & 31, wave = tid >> 5;
  const int half = lane >> 4, rsel = lane & 15;
  const bf16_t* arow = tile + (wave * 16 + rsel) * 392;
  v8f zero = {0.f, 0.f, 0.f, 0.f, 0.f, 0.f, 0.f, 0.f};
  v8f acc[NTILES];
#pragma unroll
  for (int t = 0; t < NTILES; ++t) acc[t] = zero;
  for (int kc = 0; kc < 12; ++kc) {
    v16bf a = load_a(arow, kc * 32, half);
#pragma unroll
    for (int t = 0; t < NTILES; ++t) acc[t] = wmma_bf16(a, load_b(W1p, kc, t, lane), acc[t]);
  }
#pragma unroll
  for (int t = 0; t < NTILES; ++t) {
    int n = t * 16 + rsel;
    float bv = b1[n];
#pragma unroll
    for (int i = 0; i < 8; ++i) {
      int r = wave * 16 + i + half * 8;
      hid[r * 136 + n] = f2bf(fmaxf(acc[t][i] + bv, 0.f));
    }
  }
  __syncthreads();
  const bf16_t* hrow = hid + (wave * 16 + rsel) * 136;
  v8f acc2[NTILES];
#pragma unroll
  for (int t = 0; t < NTILES; ++t) acc2[t] = zero;
  for (int kc = 0; kc < 4; ++kc) {
    v16bf a = load_a(hrow, kc * 32, half);
#pragma unroll
    for (int t = 0; t < NTILES; ++t) acc2[t] = wmma_bf16(a, load_b(W2p, kc, t, lane), acc2[t]);
  }
#pragma unroll
  for (int t = 0; t < NTILES; ++t) {
    int n = t * 16 + rsel;
    float bv = b2[n];
#pragma unroll
    for (int i = 0; i < 8; ++i) {
      long e = base + wave * 16 + i + half * 8;
      if (e < E) {
        long o = e * HD + n;
        ea[o] = ea[o] + 0.5f * (acc2[t][i] + bv);
      }
    }
  }
}

// ---------- host launcher ----------
extern "C" void kernel_launch(void* const* d_in, const int* in_sizes, int n_in,
                              void* d_out, int out_size, void* d_ws, size_t ws_size,
                              hipStream_t stream) {
  (void)n_in; (void)out_size; (void)ws_size;
  const float* x_in   = (const float*)d_in[0];
  const float* eattr  = (const float*)d_in[1];
  const float* tattr  = (const float*)d_in[2];
  const float* node_W = (const float*)d_in[3];
  const float* node_b = (const float*)d_in[4];
  const float* edge_W = (const float*)d_in[5];
  const float* edge_b = (const float*)d_in[6];
  const float* pre_W  = (const float*)d_in[7];
  const float* pre_b  = (const float*)d_in[8];
  const float* post_W = (const float*)d_in[9];
  const float* post_b = (const float*)d_in[10];
  const float* lin_W  = (const float*)d_in[11];
  const float* lin_b  = (const float*)d_in[12];
  const float* bn_g   = (const float*)d_in[13];
  const float* bn_b   = (const float*)d_in[14];
  const float* em1_W  = (const float*)d_in[15];
  const float* em1_b  = (const float*)d_in[16];
  const float* em2_W  = (const float*)d_in[17];
  const float* em2_b  = (const float*)d_in[18];
  const int*   eidx   = (const int*)d_in[19];

  const int N  = in_sizes[0] / 64;
  const int E  = in_sizes[19] / 2;
  const int ET = in_sizes[2] / 32;
  const int L  = in_sizes[8] / 128;

  float* xout   = (float*)d_out;
  float* eaout  = xout + (long)N * 128;
  float* teaout = eaout + (long)E * 128;

  char* w = (char*)d_ws;
  auto carve = [&](size_t bytes) -> char* {
    char* p = w;
    w += (bytes + 255) & ~(size_t)255;
    return p;
  };
  bf16_t*   xb     = (bf16_t*)carve((size_t)N * 128 * 2);
  float*    sum    = (float*)carve((size_t)N * 128 * 4);
  float*    sumsq  = (float*)carve((size_t)N * 128 * 4);
  unsigned* mnb    = (unsigned*)carve((size_t)N * 128 * 4);
  unsigned* mxb    = (unsigned*)carve((size_t)N * 128 * 4);
  float*    outbuf = (float*)carve((size_t)N * 128 * 4);
  float*    cnt    = (float*)carve((size_t)N * 4);
  float*    sumlog = (float*)carve(256);
  float*    musum  = (float*)carve(512);
  float*    varsum = (float*)carve(512);
  bf16_t*   nodeWp = (bf16_t*)carve((size_t)64 * 128 * 2);
  bf16_t*   edgeWp = (bf16_t*)carve((size_t)32 * 128 * 2);
  bf16_t*   preWp  = (bf16_t*)carve((size_t)L * 384 * 128 * 2);
  bf16_t*   em1Wp  = (bf16_t*)carve((size_t)L * 384 * 128 * 2);
  bf16_t*   em2Wp  = (bf16_t*)carve((size_t)L * 128 * 128 * 2);
  bf16_t*   WcP    = (bf16_t*)carve((size_t)L * 1664 * 128 * 2);
  float*    Wc32   = (float*)carve((size_t)1664 * 128 * 4);
  float*    bc     = (float*)carve((size_t)L * 128 * 4);

  // ---- weight packing ----
  k_pack<<<(64 * 128 + 255) / 256, 256, 0, stream>>>(node_W, nodeWp, 64, 0);
  k_pack<<<(32 * 128 + 255) / 256, 256, 0, stream>>>(edge_W, edgeWp, 32, 0);
  for (int l = 0; l < L; ++l) {
    k_pack<<<(384 * 128 + 255) / 256, 256, 0, stream>>>(
        pre_W + (size_t)l * 384 * 128, preWp + (size_t)l * 384 * 128, 384, 0);
    k_pack<<<(384 * 128 + 255) / 256, 256, 0, stream>>>(
        em1_W + (size_t)l * 384 * 128, em1Wp + (size_t)l * 384 * 128, 384, 1);
    k_pack<<<(128 * 128 + 255) / 256, 256, 0, stream>>>(
        em2_W + (size_t)l * 128 * 128, em2Wp + (size_t)l * 128 * 128, 128, 0);
    k_combine<<<(1664 * 128 + 255) / 256, 256, 0, stream>>>(
        post_W + (size_t)l * 1664 * 128, post_b + (size_t)l * 128,
        lin_W + (size_t)l * 128 * 128, lin_b + (size_t)l * 128, Wc32, bc + (size_t)l * 128);
    k_pack<<<(1664 * 128 + 255) / 256, 256, 0, stream>>>(
        Wc32, WcP + (size_t)l * 1664 * 128, 1664, 0);
  }

  // ---- degree statistics ----
  k_fill_u32<<<(N + 255) / 256, 256, 0, stream>>>((unsigned*)cnt, 0u, N);
  k_fill_u32<<<1, 256, 0, stream>>>((unsigned*)sumlog, 0u, 1);
  k_degree<<<(E + 255) / 256, 256, 0, stream>>>(eidx + E, cnt, E);
  k_logsum<<<(N + 255) / 256, 256, 0, stream>>>(cnt, sumlog, N);

  // ---- embeddings ----
  k_embed<<<(N + 63) / 64, 128, 0, stream>>>(x_in, 64, nodeWp, node_b, xout, xb, N);
  k_embed<<<(E + 63) / 64, 128, 0, stream>>>(eattr, 32, edgeWp, edge_b, eaout, (bf16_t*)nullptr, E);
  k_embed<<<(ET + 63) / 64, 128, 0, stream>>>(tattr, 32, edgeWp, edge_b, teaout, (bf16_t*)nullptr, ET);

  // ---- layers ----
  const long nh = (long)N * 128;
  for (int l = 0; l < L; ++l) {
    k_fill_u32<<<(int)((nh + 255) / 256), 256, 0, stream>>>((unsigned*)sum, 0u, nh);
    k_fill_u32<<<(int)((nh + 255) / 256), 256, 0, stream>>>((unsigned*)sumsq, 0u, nh);
    k_fill_u32<<<(int)((nh + 255) / 256), 256, 0, stream>>>(mnb, 0xFFFFFFFFu, nh);
    k_fill_u32<<<(int)((nh + 255) / 256), 256, 0, stream>>>(mxb, 0u, nh);
    k_edge_msg<<<(E + 63) / 64, 128, 0, stream>>>(
        xb, eaout, eidx, preWp + (size_t)l * 384 * 128, pre_b + (size_t)l * 128,
        sum, sumsq, mnb, mxb, E);
    k_node_update<<<(N + 15) / 16, 128, 0, stream>>>(
        xout, sum, sumsq, mnb, mxb, cnt, sumlog,
        WcP + (size_t)l * 1664 * 128, bc + (size_t)l * 128, outbuf, N);
    k_fill_u32<<<1, 256, 0, stream>>>((unsigned*)musum, 0u, 128);
    k_fill_u32<<<1, 256, 0, stream>>>((unsigned*)varsum, 0u, 128);
    k_bnstat<<<256, 256, 0, stream>>>(outbuf, musum, varsum, N);
    k_bnapply<<<(int)((nh + 255) / 256), 256, 0, stream>>>(
        outbuf, musum, varsum, bn_g + (size_t)l * 128, bn_b + (size_t)l * 128, xout, xb, N);
    k_edge_mlp<<<(E + 31) / 32, 64, 0, stream>>>(
        xb, eaout, eidx, em1Wp + (size_t)l * 384 * 128, em1_b + (size_t)l * 128,
        em2Wp + (size_t)l * 128 * 128, em2_b + (size_t)l * 128, E);
  }
}
